// lstm_2791728742630
// MI455X (gfx1250) — matrix operand order repaired and checked by reading, not compiled, not run on hardware
//
#include <hip/hip_runtime.h>
#include <cmath>

typedef _Float16 f16;
typedef _Float16 v16h __attribute__((ext_vector_type(16)));
typedef _Float16 v8h  __attribute__((ext_vector_type(8)));
typedef float    v8f  __attribute__((ext_vector_type(8)));

#define SEQ    168
#define BATCH  4096
#define IN_D   24
#define HID    64
#define GATES  256   // 4*HID
#define OUT_D  9
#define BT     32    // batch rows per workgroup
#define THREADS 256  // 8 waves

// ---- packed-weight scratch layout (f16 elements) -----------------------
// B-fragment order: for each matrix, frag index = nt*KT + kt, each frag is
// 512 f16 stored lane-major: elem = frag*512 + lane*16 + e.
// Per-lane mapping (dense 16-bit B, 32x16): n = lane%16, k = kt*32 + (e<8 ? 8*(lane/16) + e : 16 + 8*(lane/16) + e-8).
#define OFF_IH0 0
#define SZ_IH0  (16*1*512)           // 16 n-tiles, K padded 24->32 (1 k-tile)
#define OFF_HH0 (OFF_IH0 + SZ_IH0)
#define SZ_HH0  (16*2*512)
#define OFF_IH1 (OFF_HH0 + SZ_HH0)
#define SZ_IH1  (16*2*512)
#define OFF_HH1 (OFF_IH1 + SZ_IH1)
#define SZ_HH1  (16*2*512)
#define WS_F16_TOTAL (OFF_HH1 + SZ_HH1)

// ------------------------------------------------------------------------
__global__ void lstm_prep_weights(const float* __restrict__ Wih0,
                                  const float* __restrict__ Whh0,
                                  const float* __restrict__ Wih1,
                                  const float* __restrict__ Whh1,
                                  f16* __restrict__ ws) {
  int idx = blockIdx.x * blockDim.x + threadIdx.x;
  if (idx >= WS_F16_TOTAL) return;
  const float* W; int local, Kdim, Klog, ktiles;
  if (idx < OFF_HH0)      { local = idx - OFF_IH0; W = Wih0; Kdim = IN_D; Klog = IN_D; ktiles = 1; }
  else if (idx < OFF_IH1) { local = idx - OFF_HH0; W = Whh0; Kdim = HID;  Klog = HID;  ktiles = 2; }
  else if (idx < OFF_HH1) { local = idx - OFF_IH1; W = Wih1; Kdim = HID;  Klog = HID;  ktiles = 2; }
  else                    { local = idx - OFF_HH1; W = Whh1; Kdim = HID;  Klog = HID;  ktiles = 2; }
  int frag = local >> 9;        // /512
  int rem  = local & 511;
  int lane = rem >> 4;
  int e    = rem & 15;
  int nt   = frag / ktiles;
  int kt   = frag % ktiles;
  int n = nt * 16 + (lane & 15);
  int k = kt * 32 + ((e < 8) ? ((lane >> 4) * 8 + e) : (16 + (lane >> 4) * 8 + (e - 8)));
  float v = (k < Klog) ? W[n * Kdim + k] : 0.0f;   // B(k,n) = W[n][k] (gates = h @ W^T)
  ws[idx] = (f16)v;
}

// ------------------------------------------------------------------------
__device__ __forceinline__ v8f wmma16(v16h a, v16h b, v8f c) {
  return __builtin_amdgcn_wmma_f32_16x16x32_f16(false, a, false, b, (short)0, c,
                                                false, false);
}

// A-fragment gather from row-major LDS (ISA A layout: lane half hf picks
// K runs [8*hf, 8*hf+8) and [16+8*hf, 16+8*hf+8) within a 32-wide k-tile).
__device__ __forceinline__ v16h afrag(const f16* __restrict__ row, int kbase, int hf) {
  v8h lo = *(const v8h*)(row + kbase +      8 * hf);
  v8h hi = *(const v8h*)(row + kbase + 16 + 8 * hf);
  return __builtin_shufflevector(lo, hi, 0,1,2,3,4,5,6,7,8,9,10,11,12,13,14,15);
}

// Branch-free activations: straight-line v_exp_f32 / v_rcp_f32 (and native
// v_tanh_f32 when the builtin exists) — co-issues with the WMMA/XDL pipe.
__device__ __forceinline__ float sigm(float x) {
  return __builtin_amdgcn_rcpf(1.0f + __expf(-x));
}
__device__ __forceinline__ float ftanh(float x) {
#if __has_builtin(__builtin_amdgcn_tanhf)
  return __builtin_amdgcn_tanhf(x);
#else
  // saturates correctly at +-inf of exp; no branches, no EXEC games
  return 1.0f - 2.0f * __builtin_amdgcn_rcpf(__expf(2.0f * x) + 1.0f);
#endif
}

// ------------------------------------------------------------------------
__global__ __launch_bounds__(THREADS, 1)
void lstm2_fused(const float* __restrict__ x,
                 const float* __restrict__ b_ih0, const float* __restrict__ b_hh0,
                 const float* __restrict__ b_ih1, const float* __restrict__ b_hh1,
                 const float* __restrict__ W_out, const float* __restrict__ b_out,
                 const f16* __restrict__ wf,
                 float* __restrict__ out) {
  __shared__ __align__(16) f16   lds_x [BT * 32];        // x tile, K padded to 32
  __shared__ __align__(16) f16   lds_h0[BT * HID];       // h state, f16, A-gather source
  __shared__ __align__(16) f16   lds_h1[BT * HID];
  __shared__ __align__(16) float lds_g [BT * GATES];     // gate accum scratch (32 KB)
  __shared__ float lds_b0[GATES], lds_b1[GATES];
  __shared__ float lds_wo[OUT_D * HID];
  __shared__ float lds_bo[OUT_D];

  const int tid   = threadIdx.x;
  const int lane  = tid & 31;
  const int wv    = tid >> 5;      // wave 0..7 -> owns gate n-tiles 2wv, 2wv+1
  const int hf    = lane >> 4;     // half-wave
  const int ln    = lane & 15;
  const int bbase = blockIdx.x * BT;

  // ---- one-time init -----------------------------------------------------
  lds_b0[tid] = b_ih0[tid] + b_hh0[tid];
  lds_b1[tid] = b_ih1[tid] + b_hh1[tid];
  for (int i = tid; i < OUT_D * HID; i += THREADS) lds_wo[i] = W_out[i];
  if (tid < OUT_D) lds_bo[tid] = b_out[tid];
  for (int i = tid; i < BT * 32;  i += THREADS) lds_x[i]  = (f16)0.0f;
  for (int i = tid; i < BT * HID; i += THREADS) { lds_h0[i] = (f16)0.0f; lds_h1[i] = (f16)0.0f; }

  // ---- resident weight B-fragments (kept in VGPRs for all 168 steps) ----
  v16h bih0[2], bhh0[2][2], bih1[2][2], bhh1[2][2];
  {
    const int nt0 = 2 * wv;
#pragma unroll
    for (int j = 0; j < 2; ++j) {
      const int nt = nt0 + j;
      bih0[j] = *(const v16h*)(wf + OFF_IH0 + (size_t)nt * 512 + lane * 16);
#pragma unroll
      for (int kt = 0; kt < 2; ++kt) {
        bhh0[j][kt] = *(const v16h*)(wf + OFF_HH0 + (size_t)(nt * 2 + kt) * 512 + lane * 16);
        bih1[j][kt] = *(const v16h*)(wf + OFF_IH1 + (size_t)(nt * 2 + kt) * 512 + lane * 16);
        bhh1[j][kt] = *(const v16h*)(wf + OFF_HH1 + (size_t)(nt * 2 + kt) * 512 + lane * 16);
      }
    }
  }

  // ---- per-thread cell state: row r, 8 hidden cols starting at jb --------
  const int r  = tid >> 3;          // 0..31
  const int jb = (tid & 7) * 8;     // 0..56
  float c0[8], c1[8];
#pragma unroll
  for (int q = 0; q < 8; ++q) { c0[q] = 0.0f; c1[q] = 0.0f; }

  __syncthreads();

  for (int t = 0; t < SEQ; ++t) {
    // -- phase 1: stage x[t] tile into LDS (f16, cols 24..31 stay zero) ----
    const float* xt = x + (size_t)(t * BATCH + bbase) * IN_D;
#pragma unroll
    for (int k = 0; k < 3; ++k) {
      int idx = tid + k * THREADS;             // 0..767 == 32*24
      int rr = idx / IN_D, cc = idx % IN_D;
      lds_x[rr * 32 + cc] = (f16)xt[idx];
    }
    if (t + 1 < SEQ) __builtin_prefetch(xt + (size_t)BATCH * IN_D + tid * 3, 0, 1);
    __syncthreads();

    // -- phase 2: layer-0 gates = x@Wih0^T + h0@Whh0^T  (WMMA) -------------
#pragma unroll
    for (int mt = 0; mt < 2; ++mt) {
      const f16* arx = lds_x  + (mt * 16 + ln) * 32;
      const f16* arh = lds_h0 + (mt * 16 + ln) * HID;
      v16h ax  = afrag(arx, 0,  hf);
      v16h ah0 = afrag(arh, 0,  hf);
      v16h ah1 = afrag(arh, 32, hf);
#pragma unroll
      for (int j = 0; j < 2; ++j) {
        v8f a = {0.f,0.f,0.f,0.f,0.f,0.f,0.f,0.f};
        a = wmma16(ax,  bih0[j],    a);
        a = wmma16(ah0, bhh0[j][0], a);
        a = wmma16(ah1, bhh0[j][1], a);
        const int nt = 2 * wv + j;
#pragma unroll
        for (int vi = 0; vi < 8; ++vi)
          lds_g[(mt * 16 + hf * 8 + vi) * GATES + nt * 16 + ln] = a[vi];
      }
    }
    __syncthreads();

    // -- phase 3: layer-0 elementwise (i,f,g,o), update c0, h0 -> LDS f16 --
    {
      const float* grow = lds_g + r * GATES;
      v8h hv;
#pragma unroll
      for (int q = 0; q < 8; ++q) {
        int j = jb + q;
        float gi = grow[j]        + lds_b0[j];
        float gf = grow[64  + j]  + lds_b0[64  + j];
        float gg = grow[128 + j]  + lds_b0[128 + j];
        float go = grow[192 + j]  + lds_b0[192 + j];
        float c  = sigm(gf) * c0[q] + sigm(gi) * ftanh(gg);
        c0[q] = c;
        hv[q] = (f16)(sigm(go) * ftanh(c));
      }
      *(v8h*)(lds_h0 + r * HID + jb) = hv;
    }
    __syncthreads();

    // -- phase 4: layer-1 gates = h0@Wih1^T + h1@Whh1^T  (WMMA) ------------
#pragma unroll
    for (int mt = 0; mt < 2; ++mt) {
      const f16* a0r = lds_h0 + (mt * 16 + ln) * HID;
      const f16* a1r = lds_h1 + (mt * 16 + ln) * HID;
      v16h a00 = afrag(a0r, 0,  hf);
      v16h a01 = afrag(a0r, 32, hf);
      v16h a10 = afrag(a1r, 0,  hf);
      v16h a11 = afrag(a1r, 32, hf);
#pragma unroll
      for (int j = 0; j < 2; ++j) {
        v8f a = {0.f,0.f,0.f,0.f,0.f,0.f,0.f,0.f};
        a = wmma16(a00, bih1[j][0], a);
        a = wmma16(a01, bih1[j][1], a);
        a = wmma16(a10, bhh1[j][0], a);
        a = wmma16(a11, bhh1[j][1], a);
        const int nt = 2 * wv + j;
#pragma unroll
        for (int vi = 0; vi < 8; ++vi)
          lds_g[(mt * 16 + hf * 8 + vi) * GATES + nt * 16 + ln] = a[vi];
      }
    }
    __syncthreads();

    // -- phase 5: layer-1 elementwise, update c1, h1 -> LDS f16 ------------
    {
      const float* grow = lds_g + r * GATES;
      v8h hv;
#pragma unroll
      for (int q = 0; q < 8; ++q) {
        int j = jb + q;
        float gi = grow[j]        + lds_b1[j];
        float gf = grow[64  + j]  + lds_b1[64  + j];
        float gg = grow[128 + j]  + lds_b1[128 + j];
        float go = grow[192 + j]  + lds_b1[192 + j];
        float c  = sigm(gf) * c1[q] + sigm(gi) * ftanh(gg);
        c1[q] = c;
        hv[q] = (f16)(sigm(go) * ftanh(c));
      }
      *(v8h*)(lds_h1 + r * HID + jb) = hv;
    }
    __syncthreads();

    // -- phase 6: head  out[t] = h1 @ W_out^T + b_out  (tiny, VALU) --------
#pragma unroll
    for (int rep = 0; rep < 2; ++rep) {
      int idx = tid + rep * THREADS;
      if (idx < BT * OUT_D) {
        int rr = idx / OUT_D, oo = idx % OUT_D;
        const f16*   hrow = lds_h1 + rr * HID;
        const float* wo   = lds_wo + oo * HID;
        float acc = lds_bo[oo];
#pragma unroll
        for (int k = 0; k < HID; ++k) acc += (float)hrow[k] * wo[k];
        out[((size_t)t * BATCH + bbase + rr) * OUT_D + oo] = acc;
      }
    }
    // no trailing barrier needed: next-iter LDS writes don't conflict with
    // phase-6 reads (lds_h1 is next written only in next phase 5).
  }
}

// ------------------------------------------------------------------------
extern "C" void kernel_launch(void* const* d_in, const int* in_sizes, int n_in,
                              void* d_out, int out_size, void* d_ws, size_t ws_size,
                              hipStream_t stream) {
  (void)in_sizes; (void)n_in; (void)out_size; (void)ws_size;
  const float* x    = (const float*)d_in[0];
  const float* Wih0 = (const float*)d_in[1];
  const float* Whh0 = (const float*)d_in[2];
  const float* bih0 = (const float*)d_in[3];
  const float* bhh0 = (const float*)d_in[4];
  const float* Wih1 = (const float*)d_in[5];
  const float* Whh1 = (const float*)d_in[6];
  const float* bih1 = (const float*)d_in[7];
  const float* bhh1 = (const float*)d_in[8];
  const float* Wout = (const float*)d_in[9];
  const float* bout = (const float*)d_in[10];
  f16* wf = (f16*)d_ws;   // needs WS_F16_TOTAL*2 = 112 KB of scratch

  lstm_prep_weights<<<(WS_F16_TOTAL + 255) / 256, 256, 0, stream>>>(Wih0, Whh0, Wih1, Whh1, wf);
  lstm2_fused<<<BATCH / BT, THREADS, 0, stream>>>(x, bih0, bhh0, bih1, bhh1,
                                                  Wout, bout, wf, (float*)d_out);
}
